// SIGReg_32847909879908
// MI455X (gfx1250) — matrix operand order repaired and checked by reading, not compiled, not run on hardware
//
#include <hip/hip_runtime.h>
#include <hip/hip_bf16.h>

// ---------------------------------------------------------------------------
// SIGReg statistic on MI455X (gfx1250, wave32).
// GEMM (8192x512x1024) via V_WMMA_F32_16X16X32_BF16 with bf16 hi/lo split
// (3 WMMAs) for ~fp32 accuracy; cos/sin ECF accumulation fused per tile.
// ---------------------------------------------------------------------------

typedef __bf16 bf16;
typedef __attribute__((ext_vector_type(16))) __bf16 v16bf;
typedef __attribute__((ext_vector_type(8)))  __bf16 v8bf;
typedef __attribute__((ext_vector_type(8)))  float  v8f;

#define N_TOT 8192
#define D_TOT 512
#define P_TOT 1024
#define NKNOT 17
#define DT    0.1875f   // 3/16

union FragU { v16bf v; v8bf h[2]; };

// A-fragment (16x32 MxK, bf16), row-major LDS tile with stride 32 elements.
// Lane l<16 holds row l, K={0..7,16..23}; lane l+16 holds row l, K={8..15,24..31}.
__device__ __forceinline__ v16bf load_afrag(const bf16* base, int row, int half) {
  FragU f;
  f.h[0] = *(const v8bf*)(base + row * 32 + half * 8);        // K = half*8 .. +7
  f.h[1] = *(const v8bf*)(base + row * 32 + 16 + half * 8);   // K = 16+half*8 .. +7
  return f.v;
}

// B-fragment (32x16 KxN, bf16), LDS tile stored N-major [n][k] with stride 40
// (40 bf16 = 80 B, multiple of 16 for aligned b128 loads).
// Lane l<16 holds col l, K={0..7,16..23}; lane l+16 holds col l, K={8..15,24..31}.
__device__ __forceinline__ v16bf load_bfrag(const bf16* base, int col, int half) {
  FragU f;
  f.h[0] = *(const v8bf*)(base + col * 40 + half * 8);
  f.h[1] = *(const v8bf*)(base + col * 40 + 16 + half * 8);
  return f.v;
}

__device__ __forceinline__ v8f wmma_bf16(v16bf a, v16bf b, v8f c) {
  return __builtin_amdgcn_wmma_f32_16x16x32_bf16(false, a, false, b,
                                                 (short)0, c, false, false);
}

// ---------------------------------------------------------------------------
__global__ __launch_bounds__(256) void sigreg_zero(float* __restrict__ buf, int n) {
  int i = blockIdx.x * 256 + threadIdx.x;
  if (i < n) buf[i] = 0.0f;
}

// 1/max(||A[:,p]||, 1e-8)
__global__ __launch_bounds__(256) void sigreg_colnorm(const float* __restrict__ A,
                                                      float* __restrict__ rnorm) {
  int p = blockIdx.x * 256 + threadIdx.x;
  if (p >= P_TOT) return;
  float s = 0.0f;
  for (int d = 0; d < D_TOT; ++d) {
    float v = A[(size_t)d * P_TOT + p];
    s = fmaf(v, v, s);
  }
  rnorm[p] = 1.0f / fmaxf(sqrtf(s), 1e-8f);
}

// ---------------------------------------------------------------------------
// Main fused kernel. Grid: (P/64, N/512), block = 256 threads = 8 waves.
// Block computes proj for 64 cols x 512 rows in 8 chunks of 64 rows,
// then accumulates sum_n cos/sin(proj * t_k) into global f32 accumulators.
__global__ __launch_bounds__(256) void sigreg_main(const float* __restrict__ z,
                                                   const float* __restrict__ A,
                                                   const float* __restrict__ rnorm,
                                                   float* __restrict__ cosG,
                                                   float* __restrict__ sinG) {
  __shared__ __align__(16) bf16 zh[64 * 32];
  __shared__ __align__(16) bf16 zl[64 * 32];
  __shared__ __align__(16) bf16 anh[64 * 40];
  __shared__ __align__(16) bf16 anl[64 * 40];
  __shared__ float cosBuf[64 * NKNOT];
  __shared__ float sinBuf[64 * NKNOT];
  __shared__ float rnS[64];

  const int tid  = threadIdx.x;
  const int lane = tid & 31;
  const int wv   = tid >> 5;     // 0..7
  const int mi   = wv & 3;       // M-tile (row block of 16)
  const int pg   = wv >> 2;      // 0..1 -> owns P-tiles {2pg, 2pg+1}
  const int row  = lane & 15;
  const int half = lane >> 4;
  const int p0    = blockIdx.x * 64;
  const int nbase = blockIdx.y * 512;

  for (int i = tid; i < 64 * NKNOT; i += 256) { cosBuf[i] = 0.0f; sinBuf[i] = 0.0f; }
  if (tid < 64) rnS[tid] = rnorm[p0 + tid];
  __syncthreads();

  for (int chunk = 0; chunk < 8; ++chunk) {
    const int n0 = nbase + chunk * 64;
    v8f acc0 = {}; v8f acc1 = {};

    for (int kc = 0; kc < 16; ++kc) {
      __syncthreads();
      const int kg = kc * 32;
      // Stage z tile [64 rows x 32 k], split into bf16 hi/lo.
      for (int i = tid; i < 64 * 32; i += 256) {
        int r = i >> 5, c = i & 31;
        float v = z[(size_t)(n0 + r) * D_TOT + kg + c];
        bf16 h = (bf16)v;
        zh[r * 32 + c] = h;
        zl[r * 32 + c] = (bf16)(v - (float)h);
      }
      // Stage normalized-A tile [32 k x 64 cols], transposed to [col][k].
      for (int i = tid; i < 32 * 64; i += 256) {
        int kk = i >> 6, c = i & 63;
        float v = A[(size_t)(kg + kk) * P_TOT + p0 + c] * rnS[c];
        bf16 h = (bf16)v;
        anh[c * 40 + kk] = h;
        anl[c * 40 + kk] = (bf16)(v - (float)h);
      }
      __syncthreads();

      v16bf aH = load_afrag(zh, mi * 16 + row, half);
      v16bf aL = load_afrag(zl, mi * 16 + row, half);
      {
        v16bf bH = load_bfrag(anh, (pg * 2 + 0) * 16 + row, half);
        v16bf bL = load_bfrag(anl, (pg * 2 + 0) * 16 + row, half);
        acc0 = wmma_bf16(aH, bH, acc0);
        acc0 = wmma_bf16(aH, bL, acc0);
        acc0 = wmma_bf16(aL, bH, acc0);
      }
      {
        v16bf bH = load_bfrag(anh, (pg * 2 + 1) * 16 + row, half);
        v16bf bL = load_bfrag(anl, (pg * 2 + 1) * 16 + row, half);
        acc1 = wmma_bf16(aH, bH, acc1);
        acc1 = wmma_bf16(aH, bL, acc1);
        acc1 = wmma_bf16(aL, bH, acc1);
      }
    }

    // ECF accumulation. D layout: lane n (<16) holds m=0..7 in vgpr r;
    // lane n+16 holds m=8..15. Sum over 8 rows, fold halves via shfl_xor 16.
    for (int tt = 0; tt < 2; ++tt) {
      v8f acc = tt ? acc1 : acc0;
      const int col = (pg * 2 + tt) * 16 + row;   // block-local column
      for (int k = 0; k < NKNOT; ++k) {
        const float tk = DT * (float)k;
        float cs = 0.0f, ss = 0.0f;
#pragma unroll
        for (int r = 0; r < 8; ++r) {
          float x = acc[r] * tk;
          cs += __cosf(x);
          ss += __sinf(x);
        }
        cs += __shfl_xor(cs, 16, 32);
        ss += __shfl_xor(ss, 16, 32);
        if (half == 0) {
          atomicAdd(&cosBuf[col * NKNOT + k], cs);
          atomicAdd(&sinBuf[col * NKNOT + k], ss);
        }
      }
    }
  }

  __syncthreads();
  for (int i = tid; i < 64 * NKNOT; i += 256) {
    int c = i / NKNOT, k = i % NKNOT;
    atomicAdd(&cosG[(size_t)(p0 + c) * NKNOT + k], cosBuf[i]);
    atomicAdd(&sinG[(size_t)(p0 + c) * NKNOT + k], sinBuf[i]);
  }
}

// ---------------------------------------------------------------------------
// Scalar finalize: mean over P of ((cos_mean-phi)^2 + sin_mean^2) @ (w*phi) * N
__global__ __launch_bounds__(256) void sigreg_finalize(const float* __restrict__ cosG,
                                                       const float* __restrict__ sinG,
                                                       float* __restrict__ out) {
  __shared__ float red[256];
  const float invN = 1.0f / (float)N_TOT;
  float local = 0.0f;
  for (int i = threadIdx.x; i < P_TOT * NKNOT; i += 256) {
    int k = i % NKNOT;
    float tk  = DT * (float)k;
    float phi = __expf(-0.5f * tk * tk);
    float w   = ((k == 0) || (k == NKNOT - 1)) ? DT : 2.0f * DT;
    float cm  = cosG[i] * invN - phi;
    float sm  = sinG[i] * invN;
    local += (cm * cm + sm * sm) * (w * phi);
  }
  red[threadIdx.x] = local;
  __syncthreads();
  for (int s = 128; s > 0; s >>= 1) {
    if (threadIdx.x < s) red[threadIdx.x] += red[threadIdx.x + s];
    __syncthreads();
  }
  if (threadIdx.x == 0) out[0] = red[0] * ((float)N_TOT / (float)P_TOT);
}

// ---------------------------------------------------------------------------
extern "C" void kernel_launch(void* const* d_in, const int* in_sizes, int n_in,
                              void* d_out, int out_size, void* d_ws, size_t ws_size,
                              hipStream_t stream) {
  (void)in_sizes; (void)n_in; (void)out_size; (void)ws_size;
  const float* z = (const float*)d_in[0];   // [8192, 512]
  const float* A = (const float*)d_in[1];   // [512, 1024]
  float* out = (float*)d_out;               // scalar

  float* ws   = (float*)d_ws;
  float* cosG = ws;                          // P*K = 17408 floats
  float* sinG = ws + P_TOT * NKNOT;          // 17408 floats
  float* rn   = ws + 2 * P_TOT * NKNOT;      // 1024 floats
  const int nzero = 2 * P_TOT * NKNOT + P_TOT;   // 35840 floats (~140 KB)

  sigreg_zero<<<(nzero + 255) / 256, 256, 0, stream>>>(ws, nzero);
  sigreg_colnorm<<<P_TOT / 256, 256, 0, stream>>>(A, rn);
  dim3 grid(P_TOT / 64, N_TOT / 512);              // 16 x 16 = 256 blocks
  sigreg_main<<<grid, 256, 0, stream>>>(z, A, rn, cosG, sinG);
  sigreg_finalize<<<1, 256, 0, stream>>>(cosG, sinG, out);
}
